// Flashback_21199958573596
// MI455X (gfx1250) — matrix operand order repaired and checked by reading, not compiled, not run on hardware
//
#include <hip/hip_runtime.h>
#include <hip/hip_bf16.h>
#include <math.h>

#define B_ 64
#define T_ 512
#define F_ 512
#define POIS_ 100000

typedef __attribute__((ext_vector_type(16))) __bf16 v16bf;
typedef __attribute__((ext_vector_type(8)))  float  v8f;

__device__ __forceinline__ unsigned short f2bf(float x) {
  unsigned int u = __builtin_bit_cast(unsigned int, x);
  u += 0x7fffu + ((u >> 16) & 1u);          // round-to-nearest-even
  return (unsigned short)(u >> 16);
}
__device__ __forceinline__ __bf16 bfbits(unsigned short u) {
  return __builtin_bit_cast(__bf16, u);
}
__device__ __forceinline__ v8f wmma_bf16(v16bf a, v16bf b, v8f c) {
  // (neg_a, A, neg_b, B, c_mod, C, reuse_a, reuse_b)
  return __builtin_amdgcn_wmma_f32_16x16x32_bf16(false, a, false, b, (short)0, c,
                                                 false, false);
}
// A-operand element->K order chosen by this code (lanes 0-15: K0..15, lanes 16-31: K16..31,
// matching B). e = element index within v16bf, h = lane>>4.
__device__ __forceinline__ int a_kidx(int e, int h) {
  return (h << 4) + e;
}
// B operand 32x16 as packed by this code: lanes 0-15 hold K=0..15, lanes 16-31 hold K=16..31.
__device__ __forceinline__ int b_kidx(int e, int h) { return (h << 4) + e; }

// CDNA5 async global->LDS copy (per-lane 16B), tracked by ASYNCcnt.
__device__ __forceinline__ void async_b128(unsigned lds_off, const void* gbase,
                                           unsigned goff) {
  asm volatile("global_load_async_to_lds_b128 %0, %1, %2"
               :: "v"(lds_off), "v"(goff), "s"(gbase) : "memory");
}
__device__ __forceinline__ void wait_async0() {
  asm volatile("s_wait_asynccnt 0" ::: "memory");
}

// ---------------------------------------------------------------- init/zero
__global__ void k_init(unsigned* cnt) {
  if (threadIdx.x < 8) cnt[threadIdx.x] = 0u;
}
__global__ void k_zero_h(unsigned* hp) {          // zero 2x(64x512) bf16 ping-pong
  unsigned i = blockIdx.x * blockDim.x + threadIdx.x;   // grid 64x256
  hp[i] = 0u;
  hp[i + 16384] = 0u;
}

// ---------------------------------------------------------------- embedding
__global__ void k_embed(const int* __restrict__ user, const int* __restrict__ poi,
                        const float* __restrict__ ut, const float* __restrict__ pt,
                        unsigned short* __restrict__ xbf) {
  int bt = blockIdx.x;
  size_t u = (size_t)user[bt] * F_;
  size_t p = (size_t)poi[bt] * F_;
  unsigned short* xr = xbf + (size_t)bt * F_;
  for (int f = threadIdx.x; f < F_; f += blockDim.x)
    xr[f] = f2bf(ut[u + f] + pt[p + f]);
}

// --------------------------------------------- xw = A(bf16) @ Wih^T + bih (f32)
__global__ __launch_bounds__(256) void k_gemm_xw(
    const unsigned short* __restrict__ A,   // (B*T, F) bf16
    const float* __restrict__ W,            // (F, F), row = out feature
    const float* __restrict__ bias,         // (F)
    float* __restrict__ out)                // (B*T, F)
{
  extern __shared__ unsigned short sm[];
  unsigned short* sA = sm;                  // [64][32] bf16
  unsigned short* sB = sm + 64 * 32;        // [64][32] bf16
  const int tid = threadIdx.x;
  const int lane = tid & 31, wave = tid >> 5;
  const int h = lane >> 4, l15 = lane & 15;
  const int m0 = blockIdx.x * 64;
  const int n0 = blockIdx.y * 64;
  const int mt = wave >> 1, ntp = wave & 1; // wave -> (M-tile, N-tile-pair)
  const int srow = (tid * 8) >> 5, scol = (tid * 8) & 31;

  v8f acc0 = {}; v8f acc1 = {};
  for (int kk = 0; kk < F_; kk += 32) {
    { // stage A tile 64x32 bf16 (one b128 per thread)
      const unsigned short* src = A + (size_t)(m0 + srow) * F_ + kk + scol;
      *(uint4*)(sA + srow * 32 + scol) = *(const uint4*)src;
    }
    { // stage B tile: Wih rows n0.., convert f32->bf16
      const float* src = W + (size_t)(n0 + srow) * F_ + kk + scol;
      float4 v0 = *(const float4*)src;
      float4 v1 = *(const float4*)(src + 4);
      unsigned short* d = sB + srow * 32 + scol;
      d[0]=f2bf(v0.x); d[1]=f2bf(v0.y); d[2]=f2bf(v0.z); d[3]=f2bf(v0.w);
      d[4]=f2bf(v1.x); d[5]=f2bf(v1.y); d[6]=f2bf(v1.z); d[7]=f2bf(v1.w);
    }
    __syncthreads();
    v16bf af, b0, b1;
#pragma unroll
    for (int e = 0; e < 16; ++e) {
      af[e] = bfbits(sA[(mt * 16 + l15) * 32 + a_kidx(e, h)]);
      b0[e] = bfbits(sB[((ntp * 2 + 0) * 16 + l15) * 32 + b_kidx(e, h)]);
      b1[e] = bfbits(sB[((ntp * 2 + 1) * 16 + l15) * 32 + b_kidx(e, h)]);
    }
    acc0 = wmma_bf16(af, b0, acc0);
    acc1 = wmma_bf16(af, b1, acc1);
    __syncthreads();
  }
#pragma unroll
  for (int r = 0; r < 8; ++r) {
    int m  = m0 + mt * 16 + r + 8 * h;      // C layout: M = r + 8*(lane>>4)
    int na = n0 + (ntp * 2 + 0) * 16 + l15;
    int nb = n0 + (ntp * 2 + 1) * 16 + l15;
    out[(size_t)m * F_ + na] = acc0[r] + bias[na];
    out[(size_t)m * F_ + nb] = acc1[r] + bias[nb];
  }
}

// --------------------------------- persistent RNN scan: h' = tanh(xw_t + h@Whh^T + bhh)
__global__ __launch_bounds__(256) void k_scan(
    const float* __restrict__ xw,           // (B,T,F)
    const float* __restrict__ Whh,          // (F,F), row = out feature
    const float* __restrict__ bhh,          // (F)
    unsigned short* __restrict__ hping,     // 2 x (64*512) bf16 ping-pong (zeroed)
    unsigned* __restrict__ cnt,             // monotonic grid-sync counter (zeroed)
    unsigned short* __restrict__ seq_bf,    // (B,T,F) bf16 or null
    float* __restrict__ seq_f32)            // (B,T,F) f32 or null
{
  extern __shared__ unsigned short sm[];
  unsigned short* sW = sm;                  // [32][512] Whh slice, bf16 (32KB)
  unsigned short* sH = sm + 32 * 512;       // [64][512] h broadcast, bf16 (64KB)
  const int tid = threadIdx.x;
  const int g = blockIdx.x;                 // 0..15, owns cols [g*32, g*32+32)
  const int lane = tid & 31, wave = tid >> 5;
  const int h = lane >> 4, l15 = lane & 15;

  { // stage this WG's Whh slice once (resident for the whole scan)
    int base = tid * 64;                    // 32*512 / 256 threads
    int row = base >> 9, col = base & 511;
    const float* src = Whh + (size_t)(g * 32 + row) * F_ + col;
    unsigned short* dst = sW + row * 512 + col;
    for (int i = 0; i < 64; i += 4) {
      float4 v = *(const float4*)(src + i);
      dst[i+0]=f2bf(v.x); dst[i+1]=f2bf(v.y); dst[i+2]=f2bf(v.z); dst[i+3]=f2bf(v.w);
    }
  }
  __syncthreads();

  const int mt = wave & 3, nt = wave >> 2;  // 4 M-tiles x 2 N-tiles = 8 waves
  const int ncl = nt * 16 + l15;            // local col 0..31
  const int n_glob = g * 32 + ncl;
  const float bh = bhh[n_glob];
  const unsigned shbase = (unsigned)(unsigned long long)(void*)sH;

  int cur = 0;
  for (int t = 0; t < T_; ++t) {
    // broadcast h_prev (64KB bf16) from L2 into LDS via async tensor path
    const unsigned short* hg = hping + cur * (64 * 512);
#pragma unroll
    for (int i = 0; i < 16; ++i) {
      unsigned off = (unsigned)(tid * 16 + i * 4096);
      async_b128(shbase + off, hg, off);
    }
    wait_async0();
    __syncthreads();

    v8f acc = {};
#pragma unroll 4
    for (int kk = 0; kk < F_; kk += 32) {
      v16bf af, bf;
#pragma unroll
      for (int e = 0; e < 16; ++e) {
        af[e] = bfbits(sH[(mt * 16 + l15) * 512 + kk + a_kidx(e, h)]);
        bf[e] = bfbits(sW[ncl * 512 + kk + b_kidx(e, h)]);
      }
      acc = wmma_bf16(af, bf, acc);
    }

    unsigned short* hnext = hping + (cur ^ 1) * (64 * 512);
#pragma unroll
    for (int r = 0; r < 8; ++r) {
      int b = mt * 16 + r + 8 * h;          // batch row
      size_t sidx = (size_t)b * (T_ * F_) + (size_t)t * F_ + n_glob;
      float v = acc[r] + xw[sidx] + bh;
      float hn = tanhf(v);
      unsigned short hb = f2bf(hn);
      if (seq_f32) seq_f32[sidx] = hn;
      if (seq_bf)  seq_bf[sidx]  = hb;
      hnext[b * F_ + n_glob] = hb;
    }
    __threadfence();
    __syncthreads();
    if (tid == 0) {                          // monotonic grid sync through L2
      __hip_atomic_fetch_add(cnt, 1u, __ATOMIC_ACQ_REL, __HIP_MEMORY_SCOPE_AGENT);
      unsigned target = 16u * (unsigned)(t + 1);
      while (__hip_atomic_load(cnt, __ATOMIC_ACQUIRE, __HIP_MEMORY_SCOPE_AGENT) <
             target)
        __builtin_amdgcn_s_sleep(2);
    }
    __syncthreads();
    cur ^= 1;
  }
}

// ------------------------------- last-row spatio-temporal weights + weighted sum
__global__ __launch_bounds__(256) void k_attn(
    const float* __restrict__ lat, const float* __restrict__ lon,
    const float* __restrict__ utime, const float* __restrict__ h2,
    unsigned short* __restrict__ outbf)      // (B, F) bf16
{
  __shared__ float sw[512];
  __shared__ float red[256];
  const int b = blockIdx.x, tid = threadIdx.x;
  const float p  = 3.14159265358979323846f / 180.0f;
  const float PI2 = 2.0f * 3.14159265358979323846f;
  const float lat_i = lat[b * T_ + T_ - 1];
  const float lon_i = lon[b * T_ + T_ - 1];
  const float ut_i  = utime[b * T_ + T_ - 1];
  for (int j = tid; j < T_; j += 256) {
    float latj = lat[b * T_ + j], lonj = lon[b * T_ + j];
    float a = 0.5f - 0.5f * cosf((lat_i - latj) * p)
            + cosf(lat_i * p) * cosf(latj * p) * 0.5f * (1.0f - cosf((lon_i - lonj) * p));
    a = fminf(fmaxf(a, 0.0f), 1.0f);
    float dis = (a > 0.0f) ? 12742.0f * asinf(sqrtf(a)) : 0.0f;
    float dt = fabsf(ut_i - utime[b * T_ + j]);
    sw[j] = 0.5f * (1.0f + cosf(PI2 * dt)) * expf(-0.01f * dt) *
            expf(-100.0f * dis) + 1e-10f;
  }
  __syncthreads();
  red[tid] = sw[tid] + sw[tid + 256];
  __syncthreads();
  for (int off = 128; off > 0; off >>= 1) {
    if (tid < off) red[tid] += red[tid + off];
    __syncthreads();
  }
  const float inv = 1.0f / red[0];
  for (int f = tid; f < F_; f += 256) {
    float acc = 0.0f;
    const float* col = h2 + (size_t)b * (T_ * F_) + f;
    for (int j = 0; j < T_; ++j) {
      if (j + 16 < T_) __builtin_prefetch(col + (size_t)(j + 16) * F_, 0, 0);
      acc += sw[j] * col[(size_t)j * F_];
    }
    outbf[b * F_ + f] = f2bf(acc * inv);
  }
}

// ---------------------------------------- logits: pre = outs_last @ W_lin^T + b_lin
__global__ __launch_bounds__(256) void k_logits(
    const unsigned short* __restrict__ Abf,  // (64, 512) bf16
    const float* __restrict__ Wl,            // (POIS, F)
    const float* __restrict__ bl,            // (POIS)
    float* __restrict__ out)                 // (2, B, POIS)
{
  extern __shared__ unsigned short sm[];
  unsigned short* sA = sm;                   // [64][512] bf16 (64KB)
  const int tid = threadIdx.x;
  const int lane = tid & 31, wave = tid >> 5;
  const int h = lane >> 4, l15 = lane & 15;
  for (int i = tid * 8; i < 64 * 512; i += 256 * 8)
    *(uint4*)(sA + i) = *(const uint4*)(Abf + i);
  __syncthreads();

  const int mt = wave & 3, nt = wave >> 2;
  const int n_glob = blockIdx.x * 32 + nt * 16 + l15;
  const float* wrow = Wl + (size_t)n_glob * F_;
  v8f acc = {};
  for (int kk = 0; kk < F_; kk += 32) {
    const float* src = wrow + kk + h * 16;   // 16 consecutive f32 = this lane's K
    float4 v0 = *(const float4*)(src + 0);
    float4 v1 = *(const float4*)(src + 4);
    float4 v2 = *(const float4*)(src + 8);
    float4 v3 = *(const float4*)(src + 12);
    float tmp[16] = {v0.x,v0.y,v0.z,v0.w, v1.x,v1.y,v1.z,v1.w,
                     v2.x,v2.y,v2.z,v2.w, v3.x,v3.y,v3.z,v3.w};
    v16bf af, bf;
#pragma unroll
    for (int e = 0; e < 16; ++e) {
      af[e] = bfbits(sA[(mt * 16 + l15) * 512 + kk + a_kidx(e, h)]);
      bf[e] = bfbits(f2bf(tmp[e]));          // b_kidx(e,h) == h*16+e by construction
    }
    acc = wmma_bf16(af, bf, acc);
  }
  const float bias = bl[n_glob];
#pragma unroll
  for (int r = 0; r < 8; ++r) {
    int m = mt * 16 + r + 8 * h;
    float v = acc[r] + bias;
    out[(size_t)m * POIS_ + n_glob] = v;
    out[(size_t)(B_ + m) * POIS_ + n_glob] = v;   // tuple duplicate
  }
}

// ----------------------------------------------------------------- launcher
extern "C" void kernel_launch(void* const* d_in, const int* in_sizes, int n_in,
                              void* d_out, int out_size, void* d_ws, size_t ws_size,
                              hipStream_t stream) {
  const int*   user  = (const int*)d_in[0];
  const int*   poi   = (const int*)d_in[1];
  const float* lat   = (const float*)d_in[3];
  const float* lon   = (const float*)d_in[4];
  const float* utime = (const float*)d_in[7];
  const float* utab  = (const float*)d_in[8];
  const float* ptab  = (const float*)d_in[9];
  const float* Wih   = (const float*)d_in[10];
  const float* Whh   = (const float*)d_in[11];
  const float* bih   = (const float*)d_in[12];
  const float* bhh   = (const float*)d_in[13];
  const float* W_lin = (const float*)d_in[14];
  const float* b_lin = (const float*)d_in[15];
  float* out = (float*)d_out;

  char* ws = (char*)d_ws;
  size_t off = 0;
  auto alloc = [&](size_t bytes) -> char* {
    char* p = ws + off;
    off = (off + bytes + 255) & ~(size_t)255;
    return p;
  };
  unsigned short* xbf   = (unsigned short*)alloc((size_t)B_ * T_ * F_ * 2);
  float*          xw    = (float*)         alloc((size_t)B_ * T_ * F_ * 4);
  unsigned short* h1bf  = (unsigned short*)alloc((size_t)B_ * T_ * F_ * 2);
  float*          h2f   = (float*)         alloc((size_t)B_ * T_ * F_ * 4);
  unsigned short* hping = (unsigned short*)alloc((size_t)2 * 64 * 512 * 2);
  unsigned short* outbf = (unsigned short*)alloc((size_t)B_ * F_ * 2);
  unsigned*       cnt   = (unsigned*)      alloc(256);

  const size_t lds_gemm = 2 * (64 * 32) * sizeof(unsigned short);
  const size_t lds_scan = (32 * 512 + 64 * 512) * sizeof(unsigned short);
  const size_t lds_logit = (64 * 512) * sizeof(unsigned short);

  k_init  <<<1, 256, 0, stream>>>(cnt);
  k_zero_h<<<64, 256, 0, stream>>>((unsigned*)hping);
  k_embed <<<B_ * T_, 256, 0, stream>>>(user, poi, utab, ptab, xbf);

  // layer 0
  k_gemm_xw<<<dim3(512, 8), 256, lds_gemm, stream>>>(xbf, Wih, bih, xw);
  k_scan   <<<16, 256, lds_scan, stream>>>(xw, Whh, bhh, hping, cnt + 0,
                                           h1bf, nullptr);
  k_zero_h <<<64, 256, 0, stream>>>((unsigned*)hping);
  // layer 1
  k_gemm_xw<<<dim3(512, 8), 256, lds_gemm, stream>>>(h1bf, Wih + F_ * F_,
                                                     bih + F_, xw);
  k_scan   <<<16, 256, lds_scan, stream>>>(xw, Whh + F_ * F_, bhh + F_, hping,
                                           cnt + 1, nullptr, h2f);

  k_attn  <<<B_, 256, 0, stream>>>(lat, lon, utime, h2f, outbf);
  k_logits<<<POIS_ / 32, 256, lds_logit, stream>>>(outbf, W_lin, b_lin, out);
}
